// SAGE_8744553415267
// MI455X (gfx1250) — matrix operand order repaired and checked by reading, not compiled, not run on hardware
//
#include <hip/hip_runtime.h>
#include <hip/hip_bf16.h>

#define NN 100000
#define NE 1600000
#define D  128

typedef __attribute__((ext_vector_type(16))) _Float16 v16h;
typedef __attribute__((ext_vector_type(8)))  _Float16 v8h;
typedef __attribute__((ext_vector_type(4)))  _Float16 v4h;
typedef __attribute__((ext_vector_type(8)))  float    v8f;

// ---------------- utility kernels ----------------

__global__ __launch_bounds__(256) void sage_fill0(float4* __restrict__ p, int n4) {
  int i = blockIdx.x * 256 + threadIdx.x;
  if (i < n4) p[i] = make_float4(0.f, 0.f, 0.f, 0.f);
}

__global__ __launch_bounds__(256) void sage_deg(const int* __restrict__ dst,
                                                float* __restrict__ deg, int ne) {
  int e = blockIdx.x * 256 + threadIdx.x;
  if (e < ne) unsafeAtomicAdd(&deg[dst[e]], 1.0f);
}

__global__ __launch_bounds__(256) void sage_rdeg(float* __restrict__ deg, int n) {
  int i = blockIdx.x * 256 + threadIdx.x;
  if (i < n) deg[i] = 1.0f / fmaxf(deg[i], 1.0f);
}

// f32 -> f16 conversion of the input features (done once).
__global__ __launch_bounds__(256) void sage_cvt16(const float* __restrict__ x,
                                                  _Float16* __restrict__ o, int n4) {
  int i = blockIdx.x * 256 + threadIdx.x;
  if (i >= n4) return;
  float4 v = ((const float4*)x)[i];
  v4h h = { (_Float16)v.x, (_Float16)v.y, (_Float16)v.z, (_Float16)v.w };
  *(v4h*)(o + (size_t)i * 4) = h;
}

// Convert a 128x128 f32 weight (row-major [out][in]) to
// N-major f16: Wt[n][k] = W[n][k], so WMMA B-fragments are contiguous in K.
__global__ void sage_wprep(const float* __restrict__ W, _Float16* __restrict__ Wt) {
  int k = blockIdx.x * 16 + threadIdx.x;
  int n = blockIdx.y * 16 + threadIdx.y;
  Wt[n * D + k] = (_Float16)W[n * D + k];
}

// ---------------- aggregation (scatter-add over edges) ----------------
// One wave per edge; each lane gathers 4 f16 (8B), accumulates in f32 atomics.
__global__ __launch_bounds__(256) void sage_scatter(const _Float16* __restrict__ h16,
    const int* __restrict__ src, const int* __restrict__ dst,
    float* __restrict__ msg, int ne) {
  int e = blockIdx.x * 8 + (threadIdx.x >> 5);
  if (e >= ne) return;
  int lane = threadIdx.x & 31;
  int s = src[e], d = dst[e];
  v4h hv = *(const v4h*)(h16 + (size_t)s * D + lane * 4);
  float* o = msg + (size_t)d * D + lane * 4;
  unsafeAtomicAdd(o + 0, (float)hv[0]);
  unsafeAtomicAdd(o + 1, (float)hv[1]);
  unsafeAtomicAdd(o + 2, (float)hv[2]);
  unsafeAtomicAdd(o + 3, (float)hv[3]);
}

// ---------------- fused mean + dual-GEMM + bias (+ReLU) ----------------
// A = (msg * rdeg) rows r0..r0+15 (converted to f16 while staging to LDS)
// H = h16 rows r0..r0+15 (async DMA'd to LDS)
// out = A @ WtL + H @ WtR + bias; f16 path applies ReLU and may alias h16.
__global__ __launch_bounds__(256) void sage_gemm(
    const float* __restrict__ msg, const float* __restrict__ rdeg,
    const _Float16* __restrict__ h16,
    const _Float16* __restrict__ WtL, const _Float16* __restrict__ WtR,
    const float* __restrict__ bias,
    _Float16* __restrict__ out16, float* __restrict__ out32) {
  __shared__ __align__(16) _Float16 sA[16 * D];
  __shared__ __align__(16) _Float16 sH[16 * D];
  const int tid = threadIdx.x;
  const int r0  = blockIdx.x * 16;

  // H tile (4 KB): async global->LDS DMA, one b128 chunk per thread.
  {
    const _Float16* g = h16 + (size_t)r0 * D + tid * 8;
    unsigned ldsa = (unsigned)(unsigned long long)(uintptr_t)(sH + tid * 8);
    asm volatile("global_load_async_to_lds_b128 %0, %1, off"
                 :: "v"(ldsa), "v"(g) : "memory");
  }
  // A tile: load msg f32, scale by 1/deg, convert to f16, store to LDS.
  #pragma unroll
  for (int it = 0; it < 2; ++it) {
    int i = tid + it * 256;              // 512 float4 chunks total
    int row = i >> 5;                    // 32 float4 per row
    float r = rdeg[r0 + row];
    float4 mv = *(const float4*)(msg + (size_t)(r0 + row) * D + (i & 31) * 4);
    v4h a = { (_Float16)(mv.x * r), (_Float16)(mv.y * r),
              (_Float16)(mv.z * r), (_Float16)(mv.w * r) };
    *(v4h*)(sA + i * 4) = a;
  }
#if __has_builtin(__builtin_amdgcn_s_wait_asynccnt)
  __builtin_amdgcn_s_wait_asynccnt(0);
#else
  asm volatile("s_wait_asynccnt 0" ::: "memory");
#endif
  __syncthreads();

  const int lane = tid & 31;
  const int wave = tid >> 5;
  const int n0   = wave * 16;
  const int m    = lane & 15;          // A-fragment row for this lane
  const int kb   = (lane >> 4) * 8;    // ISA A-layout: upper half-wave holds K+8

  union U { v16h v; v8h p[2]; };
  v8f c = {};

  #pragma unroll
  for (int ks = 0; ks < 4; ++ks) {
    const int k0 = ks * 32;
    U a, h;
    a.p[0] = *(const v8h*)(sA + m * D + k0 + kb);
    a.p[1] = *(const v8h*)(sA + m * D + k0 + kb + 16);
    h.p[0] = *(const v8h*)(sH + m * D + k0 + kb);
    h.p[1] = *(const v8h*)(sH + m * D + k0 + kb + 16);
    // B fragments (32x16): lane = N column, K in same order as A.
    U ul, ur;
    ul.p[0] = *(const v8h*)(WtL + (size_t)(n0 + m) * D + k0 + kb);
    ul.p[1] = *(const v8h*)(WtL + (size_t)(n0 + m) * D + k0 + kb + 16);
    ur.p[0] = *(const v8h*)(WtR + (size_t)(n0 + m) * D + k0 + kb);
    ur.p[1] = *(const v8h*)(WtR + (size_t)(n0 + m) * D + k0 + kb + 16);
    v16h bl = ul.v;
    v16h br = ur.v;
    c = __builtin_amdgcn_wmma_f32_16x16x32_f16(false, a.v, false, bl,
                                               (short)0, c, false, false);
    c = __builtin_amdgcn_wmma_f32_16x16x32_f16(false, h.v, false, br,
                                               (short)0, c, false, false);
  }

  // C/D layout: VGPR v -> row r0 + v + (lane>>4)*8, col n0 + (lane&15).
  const int ncol = n0 + (lane & 15);
  const float b  = bias[ncol];
  const int mb   = r0 + (lane >> 4) * 8;
  if (out16) {  // hidden layers: ReLU + f16 (next layer's h); may alias h16
    #pragma unroll
    for (int v = 0; v < 8; ++v) {
      float xv = fmaxf(c[v] + b, 0.0f);
      out16[(size_t)(mb + v) * D + ncol] = (_Float16)xv;
    }
  } else {      // final layer: f32, no ReLU
    #pragma unroll
    for (int v = 0; v < 8; ++v)
      out32[(size_t)(mb + v) * D + ncol] = c[v] + b;
  }
}

// ---------------- driver ----------------

extern "C" void kernel_launch(void* const* d_in, const int* in_sizes, int n_in,
                              void* d_out, int out_size, void* d_ws, size_t ws_size,
                              hipStream_t stream) {
  const float* x   = (const float*)d_in[0];
  const int*   src = (const int*)d_in[1];
  const int*   dst = (const int*)d_in[2];
  const float* Wl0 = (const float*)d_in[3];
  const float* bl0 = (const float*)d_in[4];
  const float* Wr0 = (const float*)d_in[5];
  const float* Wl1 = (const float*)d_in[6];
  const float* bl1 = (const float*)d_in[7];
  const float* Wr1 = (const float*)d_in[8];
  const float* Wl2 = (const float*)d_in[9];
  const float* bl2 = (const float*)d_in[10];
  const float* Wr2 = (const float*)d_in[11];

  // Workspace carve-up (all offsets 256B-aligned; deg last). ~77.4 MB total.
  char* p = (char*)d_ws;
  float*    msg = (float*)p;    p += (size_t)NN * D * 4;   // 51.2 MB
  _Float16* h16 = (_Float16*)p; p += (size_t)NN * D * 2;   // 25.6 MB
  _Float16* wt  = (_Float16*)p; p += (size_t)6 * D * D * 2;
  float*    deg = (float*)p;    p += (size_t)NN * 4;

  _Float16* WtL0 = wt + 0 * D * D;
  _Float16* WtR0 = wt + 1 * D * D;
  _Float16* WtL1 = wt + 2 * D * D;
  _Float16* WtR1 = wt + 3 * D * D;
  _Float16* WtL2 = wt + 4 * D * D;
  _Float16* WtR2 = wt + 5 * D * D;

  dim3 wgrid(8, 8), wblk(16, 16);
  sage_wprep<<<wgrid, wblk, 0, stream>>>(Wl0, WtL0);
  sage_wprep<<<wgrid, wblk, 0, stream>>>(Wr0, WtR0);
  sage_wprep<<<wgrid, wblk, 0, stream>>>(Wl1, WtL1);
  sage_wprep<<<wgrid, wblk, 0, stream>>>(Wr1, WtR1);
  sage_wprep<<<wgrid, wblk, 0, stream>>>(Wl2, WtL2);
  sage_wprep<<<wgrid, wblk, 0, stream>>>(Wr2, WtR2);

  // Degrees (same for all layers), then reciprocal in place.
  const int deg4 = NN / 4;
  sage_fill0<<<(deg4 + 255) / 256, 256, 0, stream>>>((float4*)deg, deg4);
  sage_deg<<<(NE + 255) / 256, 256, 0, stream>>>(dst, deg, NE);
  sage_rdeg<<<(NN + 255) / 256, 256, 0, stream>>>(deg, NN);

  const int msg4 = NN * D / 4;  // 3.2M float4
  sage_cvt16<<<(msg4 + 255) / 256, 256, 0, stream>>>(x, h16, msg4);

  // ---- layer 0: h16(=x) -> h16 (relu) ----
  sage_fill0<<<(msg4 + 255) / 256, 256, 0, stream>>>((float4*)msg, msg4);
  sage_scatter<<<NE / 8, 256, 0, stream>>>(h16, src, dst, msg, NE);
  sage_gemm<<<NN / 16, 256, 0, stream>>>(msg, deg, h16, WtL0, WtR0, bl0,
                                         h16, nullptr);
  // ---- layer 1: h16 -> h16 (relu) ----
  sage_fill0<<<(msg4 + 255) / 256, 256, 0, stream>>>((float4*)msg, msg4);
  sage_scatter<<<NE / 8, 256, 0, stream>>>(h16, src, dst, msg, NE);
  sage_gemm<<<NN / 16, 256, 0, stream>>>(msg, deg, h16, WtL1, WtR1, bl1,
                                         h16, nullptr);
  // ---- layer 2: h16 -> d_out (f32, no relu) ----
  sage_fill0<<<(msg4 + 255) / 256, 256, 0, stream>>>((float4*)msg, msg4);
  sage_scatter<<<NE / 8, 256, 0, stream>>>(h16, src, dst, msg, NE);
  sage_gemm<<<NN / 16, 256, 0, stream>>>(msg, deg, h16, WtL2, WtR2, bl2,
                                         nullptr, (float*)d_out);
}
